// GE2ELoss_24283745092323
// MI455X (gfx1250) — matrix operand order repaired and checked by reading, not compiled, not run on hardware
//
#include <hip/hip_runtime.h>
#include <hip/hip_bf16.h>

typedef __attribute__((ext_vector_type(16))) __bf16        v16bf;
typedef __attribute__((ext_vector_type(8)))  float         v8f;
typedef __attribute__((ext_vector_type(4)))  unsigned int  v4u;

union BF16x16 { v16bf v; v4u u[2]; };

#define GN 512   // speakers
#define GM 32    // utterances per speaker
#define GD 512   // embedding dim
#define EPS 1e-6f

__device__ __forceinline__ unsigned short f2bf(float f) {
    // round-to-nearest-even f32 -> bf16
    unsigned int x = __float_as_uint(f);
    unsigned int r = x + 0x7FFFu + ((x >> 16) & 1u);
    return (unsigned short)(r >> 16);
}

// ---------------------------------------------------------------------------
// Kernel 1: centroids (f32 + bf16) and centroid norms.  One block per speaker.
// ---------------------------------------------------------------------------
__global__ void ge2e_centroids(const float* __restrict__ E,
                               float* __restrict__ centF,
                               unsigned short* __restrict__ centB,
                               float* __restrict__ cnorm) {
    const int k = blockIdx.x;
    const int t = threadIdx.x;                 // 0..255, two dims each
    const float* base = E + (size_t)k * GM * GD;
    float s0 = 0.f, s1 = 0.f;
#pragma unroll
    for (int m = 0; m < GM; ++m) {
        s0 += base[m * GD + t];
        s1 += base[m * GD + t + 256];
    }
    const float c0 = s0 * (1.f / GM);
    const float c1 = s1 * (1.f / GM);
    centF[(size_t)k * GD + t]       = c0;
    centF[(size_t)k * GD + t + 256] = c1;
    centB[(size_t)k * GD + t]       = f2bf(c0);
    centB[(size_t)k * GD + t + 256] = f2bf(c1);

    __shared__ float red[256];
    red[t] = c0 * c0 + c1 * c1;
    __syncthreads();
    for (int s = 128; s > 0; s >>= 1) {
        if (t < s) red[t] += red[t + s];
        __syncthreads();
    }
    if (t == 0) cnorm[k] = sqrtf(red[0]);
}

// ---------------------------------------------------------------------------
// Kernel 2: fused GEMM (bf16 WMMA) + diagonal (exclusive centroid, exact f32)
//           + logsumexp + per-block partial loss.  16 rows per block.
// ---------------------------------------------------------------------------
__global__ void ge2e_fused(const float* __restrict__ E,
                           const float* __restrict__ wp,
                           const float* __restrict__ bp,
                           const float* __restrict__ centF,
                           const unsigned short* __restrict__ centB,
                           const float* __restrict__ cnorm,
                           float* __restrict__ partial) {
    __shared__ float          ldsF[16][516];   // f32 rows (padded: 516*4 = 129*16 B)
    __shared__ unsigned short ldsB[16][520];   // bf16 rows (padded: 520*2 = 65*16 B)
    __shared__ float          cj[GD];          // f32 centroid of this speaker
    __shared__ float          cnL[GN];         // centroid norms
    __shared__ float          red1[16][16];    // |e|^2 partials
    __shared__ float          red2[16][16];    // e.c_j partials
    __shared__ float          enormL[16];
    __shared__ float          simsameL[16];
    __shared__ float          wsum[8][16];     // per-wave exp-sum partials
    __shared__ float          rowloss[16];

    const int t     = threadIdx.x;             // 0..255
    const int rbase = blockIdx.x * 16;         // 16 rows, all same speaker
    const int spk   = rbase >> 5;              // speaker j of this row block
    const float W  = *wp;
    const float Bb = *bp;

    // ---- stage 1: stage 16 rows into LDS (f32 + bf16), load c_j and norms --
#pragma unroll
    for (int it = 0; it < 8; ++it) {
        int idx = it * 256 + t;                // 0..2047 float4 slots
        int row = idx >> 7;
        int dv  = (idx & 127) << 2;
        float4 v = *(const float4*)(E + (size_t)(rbase + row) * GD + dv);
        *(float4*)&ldsF[row][dv] = v;
        ushort4 h = make_ushort4(f2bf(v.x), f2bf(v.y), f2bf(v.z), f2bf(v.w));
        *(ushort4*)&ldsB[row][dv] = h;
    }
    if (t < 128)
        *(float4*)&cj[t * 4] = *(const float4*)(centF + (size_t)spk * GD + t * 4);
    cnL[t]       = cnorm[t];
    cnL[t + 256] = cnorm[t + 256];
    __syncthreads();

    // ---- stage 1b: per-row |e|^2 and e.c_j (f32, deterministic) -----------
    {
        const int row = t >> 4, l = t & 15;
        float ssq = 0.f, dc = 0.f;
#pragma unroll
        for (int kk = 0; kk < 32; ++kk) {
            int d = l + kk * 16;
            float x = ldsF[row][d];
            ssq += x * x;
            dc  += x * cj[d];
        }
        red1[row][l] = ssq;
        red2[row][l] = dc;
    }
    __syncthreads();
    if (t < 16) {
        float nsq = 0.f, dcj = 0.f;
#pragma unroll
        for (int l = 0; l < 16; ++l) { nsq += red1[t][l]; dcj += red2[t][l]; }
        const float enrm   = sqrtf(nsq);
        const float cn2    = cnL[spk] * cnL[spk];
        const float dotex  = (32.f * dcj - nsq) * (1.f / 31.f);
        const float exnsq  = (1024.f * cn2 - 64.f * dcj + nsq) * (1.f / 961.f);
        const float cos_s  = dotex / fmaxf(enrm * sqrtf(fmaxf(exnsq, 0.f)), EPS);
        enormL[t]   = enrm;
        simsameL[t] = W * cos_s + Bb;
    }
    __syncthreads();

    // ---- stage 2: WMMA GEMM over 512 columns + fused epilogue -------------
    const int w       = t >> 5;                // wave 0..7
    const int lane    = t & 31;
    const int halfsel = lane >> 4;             // 0 | 1
    const int l16     = lane & 15;
    const int c16     = halfsel * 8;           // A-fragment K chunk base
    const int bk      = halfsel * 16;          // unused (B loads also use c16)

    // per-row constants for this lane's 8 output rows -> registers (branchless epilogue)
    float en[8], ss[8];
#pragma unroll
    for (int v = 0; v < 8; ++v) {
        en[v] = enormL[v + halfsel * 8];
        ss[v] = simsameL[v + halfsel * 8];
    }

    float part[8];
#pragma unroll
    for (int v = 0; v < 8; ++v) part[v] = 0.f;

#pragma unroll
    for (int tt = 0; tt < 4; ++tt) {
        const int ct      = w + tt * 8;        // column tile 0..31
        const int colbase = ct * 16;
        const int col     = colbase + l16;
        const unsigned short* Bcol = centB + (size_t)col * GD;

        v8f acc;
#pragma unroll
        for (int v = 0; v < 8; ++v) acc[v] = 0.f;

#pragma unroll 4
        for (int k0 = 0; k0 < GD; k0 += 32) {
            BF16x16 A, Bv;
            // A (16x32 bf16): lane holds M=l16, K = c16+{0..7} and c16+{16..23}
            A.u[0] = *(const v4u*)&ldsB[l16][k0 + c16];
            A.u[1] = *(const v4u*)&ldsB[l16][k0 + c16 + 16];
            // B (32x16 bf16): lane holds N=l16, K = c16+{0..7} and c16+{16..23}
            Bv.u[0] = *(const v4u*)&Bcol[k0 + c16];
            Bv.u[1] = *(const v4u*)&Bcol[k0 + c16 + 16];
            acc = __builtin_amdgcn_wmma_f32_16x16x32_bf16(
                false, A.v, false, Bv.v, (short)0, acc, false, false);
        }

        // epilogue: dot -> sim -> exp; diagonal col replaced via cndmask,
        // division replaced by v_rcp_f32 (~1 ulp, plenty for cosine sims)
        const float cnv    = cnL[col];
        const bool  isdiag = (col == spk);
#pragma unroll
        for (int v = 0; v < 8; ++v) {
            const float denom = fmaxf(en[v] * cnv, EPS);
            float s = fmaf(acc[v] * __builtin_amdgcn_rcpf(denom), W, Bb);
            s = isdiag ? ss[v] : s;
            part[v] += __expf(s);
        }
    }

    // ---- wave-level deterministic reduction of exp-sums -------------------
#pragma unroll
    for (int v = 0; v < 8; ++v) {
        float x = part[v];
        x += __shfl_xor(x, 1, 32);
        x += __shfl_xor(x, 2, 32);
        x += __shfl_xor(x, 4, 32);
        x += __shfl_xor(x, 8, 32);             // sum within each 16-lane half
        if (lane == 0)  wsum[w][v]     = x;    // rows 0..7
        if (lane == 16) wsum[w][v + 8] = x;    // rows 8..15
    }
    __syncthreads();

    if (t < 16) {
        float es = 0.f;
#pragma unroll
        for (int ww = 0; ww < 8; ++ww) es += wsum[ww][t];
        const float lse = logf(es + EPS);
        rowloss[t] = lse - simsameL[t];        // -(diag - lse)
    }
    __syncthreads();
    if (t == 0) {
        float s = 0.f;
#pragma unroll
        for (int r = 0; r < 16; ++r) s += rowloss[r];
        partial[blockIdx.x] = s;
    }
}

// ---------------------------------------------------------------------------
// Kernel 3: deterministic final reduction of 1024 block partials.
// ---------------------------------------------------------------------------
__global__ void ge2e_finalize(const float* __restrict__ partial,
                              float* __restrict__ out) {
    __shared__ float red[256];
    const int t = threadIdx.x;
    red[t] = partial[t] + partial[t + 256] + partial[t + 512] + partial[t + 768];
    __syncthreads();
    for (int s = 128; s > 0; s >>= 1) {
        if (t < s) red[t] += red[t + s];
        __syncthreads();
    }
    if (t == 0) out[0] = red[0];
}

// ---------------------------------------------------------------------------
extern "C" void kernel_launch(void* const* d_in, const int* in_sizes, int n_in,
                              void* d_out, int out_size, void* d_ws, size_t ws_size,
                              hipStream_t stream) {
    const float* E  = (const float*)d_in[0];   // [512, 32, 512] f32
    const float* wp = (const float*)d_in[1];   // scalar
    const float* bp = (const float*)d_in[2];   // scalar
    float* out = (float*)d_out;

    // workspace layout
    float*          centF   = (float*)d_ws;                       // 512*512 f32  (1 MB)
    unsigned short* centB   = (unsigned short*)(centF + GN * GD); // 512*512 bf16 (512 KB)
    float*          cnorm   = (float*)(centB + GN * GD);          // 512 f32
    float*          partial = cnorm + GN;                         // 1024 f32

    ge2e_centroids<<<GN, 256, 0, stream>>>(E, centF, centB, cnorm);
    ge2e_fused<<<(GN * GM) / 16, 256, 0, stream>>>(E, wp, bp, centF, centB, cnorm, partial);
    ge2e_finalize<<<1, 256, 0, stream>>>(partial, out);
}
